// GCN_NP_44272522887509
// MI455X (gfx1250) — matrix operand order repaired and checked by reading, not compiled, not run on hardware
//
#include <hip/hip_runtime.h>
#include <hip/hip_bf16.h>

// ---------------------------------------------------------------------------
// GCN 2-layer forward for gfx1250 (MI455X), wave32.
//   E1 = emb @ W1 (256x256x256, f32)         -- layer-1 GEMM collapsed
//   out1[dst] += E1[x[src]] * norm           -- edge scatter (incl. self loops)
//   h1 = relu(out1 + b1) -> split bf16 hi/lo
//   hw2 = h1 @ W2 via v_wmma_f32_16x16x32_bf16 (3-term split for f32 fidelity)
//   out2[dst] += hw2[src] * norm ; + b2
//   d_out = log_softmax(out2[mask])
// ---------------------------------------------------------------------------

typedef __attribute__((ext_vector_type(16))) __bf16 v16bf;
typedef __attribute__((ext_vector_type(8)))  __bf16 v8bf;
typedef __attribute__((ext_vector_type(8)))  float  v8f;

#define DIM 256

// ---------------- degree / norm ----------------
__global__ void init_deg(float* deg, int n) {
  int i = blockIdx.x * blockDim.x + threadIdx.x;
  if (i < n) deg[i] = 1.0f;                       // self loop
}

__global__ void count_deg(const int* __restrict__ dst, float* deg, int ne) {
  int e = blockIdx.x * blockDim.x + threadIdx.x;
  if (e < ne) atomicAdd(deg + dst[e], 1.0f);
}

__global__ void finish_dinv(float* deg, int n) {
  int i = blockIdx.x * blockDim.x + threadIdx.x;
  if (i < n) {
    float d = deg[i];
    deg[i] = d > 0.0f ? rsqrtf(d) : 0.0f;
  }
}

// ---------------- tiny f32 GEMM: E1 = emb @ W1 ----------------
__global__ void gemm_E1(const float* __restrict__ emb,
                        const float* __restrict__ W1,
                        float* __restrict__ E1) {
  int c = threadIdx.x;   // 0..255
  int r = blockIdx.x;    // 0..255
  const float* e = emb + r * DIM;
  float s = 0.0f;
  #pragma unroll 8
  for (int k = 0; k < DIM; ++k) s = fmaf(e[k], W1[k * DIM + c], s);
  E1[r * DIM + c] = s;
}

// ---------------- W2 -> transposed split-bf16 ----------------
__global__ void convert_W2t(const float* __restrict__ W2,
                            __bf16* __restrict__ hi, __bf16* __restrict__ lo) {
  int c = threadIdx.x;   // col n
  int k = blockIdx.x;    // row k
  float v = W2[k * DIM + c];
  __bf16 h = (__bf16)v;
  hi[c * DIM + k] = h;
  lo[c * DIM + k] = (__bf16)(v - (float)h);
}

// ---------------- bias broadcast init ----------------
__global__ void init_bias(float* __restrict__ out, const float* __restrict__ bias, int n) {
  int i = blockIdx.x * blockDim.x + threadIdx.x;
  if (i < n) out[i] = bias[i & (DIM - 1)];
}

// ---------------- edge scatter: out[dst] += feat[row(src)] * norm ----------------
// One wave per (virtual) edge; virtual edges >= n_edges are self loops.
template <bool GATHER>
__global__ void __launch_bounds__(256) edge_agg(
    const int* __restrict__ src, const int* __restrict__ dst,
    const int* __restrict__ xidx, const float* __restrict__ dinv,
    const float* __restrict__ feat, float* __restrict__ out,
    int n_edges, int n_nodes) {
  int e    = blockIdx.x * 8 + (threadIdx.x >> 5);
  int lane = threadIdx.x & 31;
  int total = n_edges + n_nodes;
  if (e >= total) return;
  int s, d;
  if (e < n_edges) { s = src[e]; d = dst[e]; }
  else             { s = e - n_edges; d = s; }
  float w = dinv[s] * dinv[d];
  int row = GATHER ? xidx[s] : s;
  const float4* frow = (const float4*)(feat + (size_t)row * DIM);
  float*        orow = out + (size_t)d * DIM;
  float4 v0 = frow[lane];
  float4 v1 = frow[lane + 32];
  int c0 = lane * 4, c1 = (lane + 32) * 4;
  atomicAdd(orow + c0 + 0, v0.x * w);
  atomicAdd(orow + c0 + 1, v0.y * w);
  atomicAdd(orow + c0 + 2, v0.z * w);
  atomicAdd(orow + c0 + 3, v0.w * w);
  atomicAdd(orow + c1 + 0, v1.x * w);
  atomicAdd(orow + c1 + 1, v1.y * w);
  atomicAdd(orow + c1 + 2, v1.z * w);
  atomicAdd(orow + c1 + 3, v1.w * w);
}

// ---------------- relu + split into bf16 hi/lo ----------------
__global__ void relu_split(const float* __restrict__ in,
                           __bf16* __restrict__ hi, __bf16* __restrict__ lo, int n) {
  int i = blockIdx.x * blockDim.x + threadIdx.x;
  if (i >= n) return;
  float v = in[i];
  v = v > 0.0f ? v : 0.0f;
  __bf16 h = (__bf16)v;
  hi[i] = h;
  lo[i] = (__bf16)(v - (float)h);
}

// ---------------- layer-2 GEMM: C(Mx256) = A(Mx256) @ W2(256x256) ----------------
// Split-bf16 emulated f32: A=Ahi+Alo, B=Bhi+Blo; acc += AhiBhi + AhiBlo + AloBhi.
// One wave per 16x16 C tile; K=256 in 8 chunks of 32.
__global__ void __launch_bounds__(256) gemm_split_bf16(
    const __bf16* __restrict__ Ahi, const __bf16* __restrict__ Alo,
    const __bf16* __restrict__ Bhi, const __bf16* __restrict__ Blo,
    float* __restrict__ C, int M) {
  int wave   = blockIdx.x * 8 + (threadIdx.x >> 5);
  int lane   = threadIdx.x & 31;
  int tile_m = wave >> 4;          // 256/16 = 16 col tiles
  int tile_n = wave & 15;
  if (tile_m * 16 >= M) return;
  int half = lane >> 4;
  int l16  = lane & 15;

  int arow_idx = tile_m * 16 + l16;
  if (arow_idx >= M) arow_idx = M - 1;            // safe dup (M%16==0 normally)
  const __bf16* arow_hi = Ahi + (size_t)arow_idx * DIM;
  const __bf16* arow_lo = Alo + (size_t)arow_idx * DIM;
  const __bf16* brow_hi = Bhi + (size_t)(tile_n * 16 + l16) * DIM;  // W2 col as row
  const __bf16* brow_lo = Blo + (size_t)(tile_n * 16 + l16) * DIM;

  v8f acc = {};
  #pragma unroll
  for (int kc = 0; kc < DIM; kc += 32) {
    // A 16x32 bf16 layout: half h holds K {8h..8h+7} and {16+8h..16+8h+7}
    v8bf a0h = *(const v8bf*)(arow_hi + kc + half * 8);
    v8bf a1h = *(const v8bf*)(arow_hi + kc + 16 + half * 8);
    v8bf a0l = *(const v8bf*)(arow_lo + kc + half * 8);
    v8bf a1l = *(const v8bf*)(arow_lo + kc + 16 + half * 8);
    // B 32x16 bf16 layout: half h holds K {8h..8h+7} and {16+8h..16+8h+7} for column n
    v8bf b0h = *(const v8bf*)(brow_hi + kc + half * 8);
    v8bf b1h = *(const v8bf*)(brow_hi + kc + 16 + half * 8);
    v8bf b0l = *(const v8bf*)(brow_lo + kc + half * 8);
    v8bf b1l = *(const v8bf*)(brow_lo + kc + 16 + half * 8);
    v16bf a_hi, a_lo, b_hi, b_lo;
    #pragma unroll
    for (int j = 0; j < 8; ++j) {
      a_hi[j] = a0h[j]; a_hi[8 + j] = a1h[j];
      a_lo[j] = a0l[j]; a_lo[8 + j] = a1l[j];
      b_hi[j] = b0h[j]; b_hi[8 + j] = b1h[j];
      b_lo[j] = b0l[j]; b_lo[8 + j] = b1l[j];
    }
    acc = __builtin_amdgcn_wmma_f32_16x16x32_bf16(false, a_hi, false, b_hi,
                                                  (short)0, acc, false, false);
    acc = __builtin_amdgcn_wmma_f32_16x16x32_bf16(false, a_hi, false, b_lo,
                                                  (short)0, acc, false, false);
    acc = __builtin_amdgcn_wmma_f32_16x16x32_bf16(false, a_lo, false, b_hi,
                                                  (short)0, acc, false, false);
  }

  // C/D layout: VGPR r -> row tile_m*16 + 8*half + r, col tile_n*16 + l16
  int crow0 = tile_m * 16 + half * 8;
  int ccol  = tile_n * 16 + l16;
  #pragma unroll
  for (int r = 0; r < 8; ++r) {
    int rr = crow0 + r;
    if (rr < M) C[(size_t)rr * DIM + ccol] = acc[r];
  }
}

// ---------------- log_softmax on masked rows (one wave per row) ----------------
__global__ void __launch_bounds__(256) logsoftmax_rows(
    const float* __restrict__ h, const int* __restrict__ mask,
    float* __restrict__ y, int n_mask) {
  int r    = blockIdx.x * 8 + (threadIdx.x >> 5);
  int lane = threadIdx.x & 31;
  if (r >= n_mask) return;
  const float* row = h + (size_t)mask[r] * DIM;
  float v[8];
  float mx = -3.402823466e38f;
  #pragma unroll
  for (int j = 0; j < 8; ++j) { v[j] = row[lane + 32 * j]; mx = fmaxf(mx, v[j]); }
  #pragma unroll
  for (int off = 16; off; off >>= 1) mx = fmaxf(mx, __shfl_xor(mx, off, 32));
  float s = 0.0f;
  #pragma unroll
  for (int j = 0; j < 8; ++j) s += expf(v[j] - mx);
  #pragma unroll
  for (int off = 16; off; off >>= 1) s += __shfl_xor(s, off, 32);
  float lse = mx + logf(s);
  float* yr = y + (size_t)r * DIM;
  #pragma unroll
  for (int j = 0; j < 8; ++j) yr[lane + 32 * j] = v[j] - lse;
}

// ---------------------------------------------------------------------------
extern "C" void kernel_launch(void* const* d_in, const int* in_sizes, int n_in,
                              void* d_out, int out_size, void* d_ws, size_t ws_size,
                              hipStream_t stream) {
  const int*   x    = (const int*)d_in[0];
  const int*   ei   = (const int*)d_in[1];
  const int*   mask = (const int*)d_in[2];
  const float* emb  = (const float*)d_in[3];
  const float* W1   = (const float*)d_in[4];
  const float* b1   = (const float*)d_in[5];
  const float* W2   = (const float*)d_in[6];
  const float* b2   = (const float*)d_in[7];
  (void)n_in; (void)out_size; (void)ws_size;

  const int n_nodes = in_sizes[0];        // 100000
  const int n_edges = in_sizes[1] / 2;    // 800000
  const int n_mask  = in_sizes[2];        // 10000
  const int* src = ei;
  const int* dst = ei + n_edges;
  const int feat_n = n_nodes * DIM;       // 25,600,000

  // ---- workspace carve-out (all 512B aligned) ----
  char*  ws  = (char*)d_ws;
  size_t off = 0;
  auto take = [&](size_t bytes) -> char* {
    char* p = ws + off;
    off += (bytes + 511) & ~(size_t)511;
    return p;
  };
  float*  dinv   = (float*)take((size_t)n_nodes * 4);        // deg -> dinv in place
  float*  E1     = (float*)take((size_t)DIM * DIM * 4);
  __bf16* W2t_hi = (__bf16*)take((size_t)DIM * DIM * 2);
  __bf16* W2t_lo = (__bf16*)take((size_t)DIM * DIM * 2);
  float*  bufA   = (float*)take((size_t)feat_n * 4);         // out1, then hw2
  char*   bufB   = take((size_t)feat_n * 4);                 // h1 hi|lo, then out2
  __bf16* h1hi   = (__bf16*)bufB;
  __bf16* h1lo   = (__bf16*)(bufB + (size_t)feat_n * 2);
  float*  out2   = (float*)bufB;

  const int T = 256;
  int nblk  = (n_nodes + T - 1) / T;
  int eblk  = (n_edges + T - 1) / T;
  int fblk  = (feat_n + T - 1) / T;
  int vedges = n_edges + n_nodes;                 // real + self loops
  int agblk = (vedges + 7) / 8;                   // 8 waves / block, wave per edge
  int tiles  = ((n_nodes + 15) / 16) * (DIM / 16);
  int gmblk  = (tiles + 7) / 8;
  int lsblk  = (n_mask + 7) / 8;

  // degrees -> dinv
  init_deg   <<<nblk, T, 0, stream>>>(dinv, n_nodes);
  count_deg  <<<eblk, T, 0, stream>>>(dst, dinv, n_edges);
  finish_dinv<<<nblk, T, 0, stream>>>(dinv, n_nodes);

  // small dense precomputes
  gemm_E1    <<<DIM, DIM, 0, stream>>>(emb, W1, E1);
  convert_W2t<<<DIM, DIM, 0, stream>>>(W2, W2t_hi, W2t_lo);

  // layer 1: out1 = b1 + scatter(E1[x[src]] * norm)
  init_bias  <<<fblk, T, 0, stream>>>(bufA, b1, feat_n);
  edge_agg<true><<<agblk, T, 0, stream>>>(src, dst, x, dinv, E1, bufA,
                                          n_edges, n_nodes);
  relu_split <<<fblk, T, 0, stream>>>(bufA, h1hi, h1lo, feat_n);

  // layer 2 GEMM: hw2 (reusing bufA) = h1 @ W2   [WMMA bf16-split]
  gemm_split_bf16<<<gmblk, T, 0, stream>>>(h1hi, h1lo, W2t_hi, W2t_lo,
                                           bufA, n_nodes);

  // layer 2 scatter: out2 = b2 + scatter(hw2[src] * norm)
  init_bias  <<<fblk, T, 0, stream>>>(out2, b2, feat_n);
  edge_agg<false><<<agblk, T, 0, stream>>>(src, dst, nullptr, dinv, bufA, out2,
                                           n_edges, n_nodes);

  // masked log_softmax
  logsoftmax_rows<<<lsblk, T, 0, stream>>>(out2, mask, (float*)d_out, n_mask);
}
